// VoxelNeXtHead_70480413327748
// MI455X (gfx1250) — matrix operand order repaired and checked by reading, not compiled, not run on hardware
//
#include <hip/hip_runtime.h>
#include <hip/hip_bf16.h>
#include <math.h>

// ---------------- constants (match reference) ----------------
#define BATCH   4
#define NVOX    32768
#define NTOT    (BATCH * NVOX)   // 131072
#define CDIM    128
#define NCLASS  10
#define KTOP    500
#define NCAND   (NCLASS * KTOP)  // 5000

typedef __bf16 bf16_t;
typedef __attribute__((ext_vector_type(16))) __bf16 v16bf;
typedef __attribute__((ext_vector_type(8)))  float  v8f;

union Frag16 { v16bf v; uint4 q[2]; };

// ---------------- workspace layout (bytes) ----------------
#define OFF_BPHI   0          // [6][128][128] bf16, W1^T hi      (196608)
#define OFF_BPLO   196608     // [6][128][128] bf16, W1^T lo      (196608)
#define OFF_BNA    393216     // [6][128] f32 BN scale            (3072)
#define OFF_BNB    396288     // [6][128] f32 BN shift (b1 folded)(3072)
#define OFF_SCORES 399360     // [NTOT][10] f32 sigmoid heatmap   (5242880)
#define OFF_REG    5642240    // [NTOT][10] f32 regression        (5242880)
#define OFF_CSC    10885120   // [B][10][500] f32 cand scores     (80000)
#define OFF_CIX    10965120   // [B][10][500] u32 cand voxel idx  (80000)

// =====================================================================
// Kernel 1: split W1 into bf16 hi/lo (transposed for WMMA-B lane loads)
// and fold bias + BN(eval) into per-channel scale/shift.
// =====================================================================
__global__ __launch_bounds__(256)
void prep_kernel(const float* __restrict__ W1, const float* __restrict__ b1,
                 const float* __restrict__ gma, const float* __restrict__ bta,
                 const float* __restrict__ mu,  const float* __restrict__ var,
                 bf16_t* __restrict__ BpHi, bf16_t* __restrict__ BpLo,
                 float* __restrict__ bnA, float* __restrict__ bnB) {
  int e = blockIdx.x * 256 + threadIdx.x;
  if (e < 6 * CDIM * CDIM) {
    int i   = e >> 14;          // head
    int rem = e & 16383;
    int c   = rem >> 7;         // input channel (K)
    int d   = rem & 127;        // output channel (N)
    float w = W1[e];            // W1[i][c][d]
    bf16_t hi = (bf16_t)w;
    bf16_t lo = (bf16_t)(w - (float)hi);
    int o = (i * CDIM + d) * CDIM + c;   // transposed: [i][d][c]
    BpHi[o] = hi;
    BpLo[o] = lo;
  }
  if (e < 6 * CDIM) {
    float inv = gma[e] / sqrtf(var[e] + 1e-5f);
    bnA[e] = inv;
    bnB[e] = (b1[e] - mu[e]) * inv + bta[e];
  }
}

// =====================================================================
// Kernel 2: fused 6-head GEMM (split-bf16 WMMA) + BN + ReLU + small
// output projections. One WG per 128-voxel row tile, 8 waves of 32.
// Per-wave tile: 64 rows x 32 cols (4 row-tiles x 2 col-tiles) so each
// B fragment load is amortized over 4 row-tiles (4x less L2 traffic).
// =====================================================================
template <int OC>
__device__ __forceinline__ void proj_row(const float* __restrict__ hrow,
                                         const float* __restrict__ w2,
                                         const float* __restrict__ b2,
                                         float* __restrict__ acc) {
#pragma unroll
  for (int j = 0; j < OC; ++j) acc[j] = b2[j];
  for (int c = 0; c < CDIM; ++c) {
    float hv = hrow[c];
#pragma unroll
    for (int j = 0; j < OC; ++j) acc[j] = fmaf(hv, w2[c * OC + j], acc[j]);
  }
}

__global__ __launch_bounds__(256)
void head_gemm_kernel(const float* __restrict__ feats,
                      const bf16_t* __restrict__ BpHi, const bf16_t* __restrict__ BpLo,
                      const float* __restrict__ bnA, const float* __restrict__ bnB,
                      const float* __restrict__ W2hm, const float* __restrict__ b2hm,
                      const float* __restrict__ W2ct, const float* __restrict__ b2ct,
                      const float* __restrict__ W2cz, const float* __restrict__ b2cz,
                      const float* __restrict__ W2dm, const float* __restrict__ b2dm,
                      const float* __restrict__ W2rt, const float* __restrict__ b2rt,
                      const float* __restrict__ W2vl, const float* __restrict__ b2vl,
                      float* __restrict__ scoresOut, float* __restrict__ regOut) {
  __shared__ __align__(16) bf16_t Ahi[128][136];   // feats tile, bf16 hi
  __shared__ __align__(16) bf16_t Alo[128][136];   // feats tile, bf16 lo (residual)
  __shared__ __align__(16) float  Hs[128][133];    // per-head hidden activations

  const int tid  = threadIdx.x;
  const int wave = tid >> 5;
  const int lane = tid & 31;
  const int base = blockIdx.x * 128;               // first voxel row of tile

  // ---- stage feats tile into LDS as split bf16 (vectorized) ----
  for (int idx = tid; idx < 128 * 32; idx += 256) {   // 16 iterations, float4 each
    int r  = idx >> 5;
    int c4 = (idx & 31) * 4;
    float4 f = *(const float4*)&feats[(size_t)(base + r) * CDIM + c4];
    float fs[4] = {f.x, f.y, f.z, f.w};
    union { bf16_t h[4]; unsigned long long u; } ph, pl;
#pragma unroll
    for (int j = 0; j < 4; ++j) {
      bf16_t hi = (bf16_t)fs[j];
      ph.h[j] = hi;
      pl.h[j] = (bf16_t)(fs[j] - (float)hi);
    }
    *(unsigned long long*)&Ahi[r][c4] = ph.u;
    *(unsigned long long*)&Alo[r][c4] = pl.u;
  }
  __syncthreads();

  const float* W2s[6] = {W2hm, W2ct, W2cz, W2dm, W2rt, W2vl};
  const float* b2s[6] = {b2hm, b2ct, b2cz, b2dm, b2rt, b2vl};

  const int rowGroup = wave >> 2;     // 0..1 -> 64 rows each
  const int colGroup = wave & 3;      // 0..3 -> 32 cols each

  for (int i = 0; i < 6; ++i) {                    // six heads
    const bf16_t* Bh = BpHi + (size_t)i * CDIM * CDIM;
    const bf16_t* Bl = BpLo + (size_t)i * CDIM * CDIM;

    // warm L2 with next head's B while this head computes
    if (i < 5) {
      __builtin_prefetch(Bh + CDIM * CDIM + tid * 64, 0, 0);
      __builtin_prefetch(Bl + CDIM * CDIM + tid * 64, 0, 0);
    }

    v8f acc[4][2];
#pragma unroll
    for (int rt = 0; rt < 4; ++rt)
#pragma unroll
      for (int cc = 0; cc < 2; ++cc)
#pragma unroll
        for (int v = 0; v < 8; ++v) acc[rt][cc][v] = 0.0f;

#pragma unroll
    for (int kk = 0; kk < 4; ++kk) {               // K = 128 in steps of 32
      // B fragments (16-bit B 32x16): lane n holds contiguous K run for col n.
      const int kbase = kk * 32 + ((lane >> 4) ? 16 : 0);
      Frag16 bhi[2], blo[2];
#pragma unroll
      for (int cc = 0; cc < 2; ++cc) {
        const int d = (colGroup * 2 + cc) * 16 + (lane & 15);
        const uint4* ph = (const uint4*)(Bh + d * CDIM + kbase);
        bhi[cc].q[0] = ph[0]; bhi[cc].q[1] = ph[1];
        const uint4* pl = (const uint4*)(Bl + d * CDIM + kbase);
        blo[cc].q[0] = pl[0]; blo[cc].q[1] = pl[1];
      }

      // A fragments (16-bit A 16x32): lanes<16 -> K {0..15},
      // lanes>=16 -> K {16..31}, row = lane&15.
      const int koff = kk * 32 + ((lane >> 4) ? 16 : 0);
#pragma unroll
      for (int rt = 0; rt < 4; ++rt) {
        const int arow = rowGroup * 64 + rt * 16 + (lane & 15);
        Frag16 ahi, alo;
        ahi.q[0] = *(const uint4*)&Ahi[arow][koff];
        ahi.q[1] = *(const uint4*)&Ahi[arow][koff + 8];
        alo.q[0] = *(const uint4*)&Alo[arow][koff];
        alo.q[1] = *(const uint4*)&Alo[arow][koff + 8];
#pragma unroll
        for (int cc = 0; cc < 2; ++cc) {
          // split-bf16 ~fp32: hi*hi + lo*hi + hi*lo
          acc[rt][cc] = __builtin_amdgcn_wmma_f32_16x16x32_bf16(
              false, ahi.v, false, bhi[cc].v, (short)0, acc[rt][cc], false, false);
          acc[rt][cc] = __builtin_amdgcn_wmma_f32_16x16x32_bf16(
              false, alo.v, false, bhi[cc].v, (short)0, acc[rt][cc], false, false);
          acc[rt][cc] = __builtin_amdgcn_wmma_f32_16x16x32_bf16(
              false, ahi.v, false, blo[cc].v, (short)0, acc[rt][cc], false, false);
        }
      }
    }

    // ---- BN + ReLU on C fragments, spill to LDS ----
#pragma unroll
    for (int cc = 0; cc < 2; ++cc) {
      const int col = (colGroup * 2 + cc) * 16 + (lane & 15);
      const float a = bnA[i * CDIM + col];
      const float b = bnB[i * CDIM + col];
#pragma unroll
      for (int rt = 0; rt < 4; ++rt) {
#pragma unroll
        for (int v = 0; v < 8; ++v) {
          const int r = rowGroup * 64 + rt * 16 + v + ((lane >> 4) ? 8 : 0);
          float x = fmaf(acc[rt][cc][v], a, b);
          Hs[r][col] = x > 0.0f ? x : 0.0f;
        }
      }
    }
    __syncthreads();

    // ---- tiny output projection (one thread per voxel row) ----
    if (tid < 128) {
      const size_t g = (size_t)(base + tid);
      const float* hrow = &Hs[tid][0];
      float acc2[10];
      switch (i) {
        case 0:
          proj_row<10>(hrow, W2s[0], b2s[0], acc2);
          for (int j = 0; j < 10; ++j)
            scoresOut[g * 10 + j] = 1.0f / (1.0f + expf(-acc2[j]));
          break;
        case 1:
          proj_row<2>(hrow, W2s[1], b2s[1], acc2);
          regOut[g * 10 + 0] = acc2[0]; regOut[g * 10 + 1] = acc2[1];
          break;
        case 2:
          proj_row<1>(hrow, W2s[2], b2s[2], acc2);
          regOut[g * 10 + 2] = acc2[0];
          break;
        case 3:
          proj_row<3>(hrow, W2s[3], b2s[3], acc2);
          regOut[g * 10 + 3] = expf(acc2[0]);
          regOut[g * 10 + 4] = expf(acc2[1]);
          regOut[g * 10 + 5] = expf(acc2[2]);
          break;
        case 4:
          proj_row<2>(hrow, W2s[4], b2s[4], acc2);
          regOut[g * 10 + 6] = acc2[0]; regOut[g * 10 + 7] = acc2[1];
          break;
        default:
          proj_row<2>(hrow, W2s[5], b2s[5], acc2);
          regOut[g * 10 + 8] = acc2[0]; regOut[g * 10 + 9] = acc2[1];
          break;
      }
    }
    __syncthreads();   // Hs reused by next head
  }
}

// =====================================================================
// Bitonic sort (descending) over LDS keys. All threads must call.
// =====================================================================
__device__ void bitonic_desc(unsigned long long* keys, unsigned n,
                             unsigned tid, unsigned nth) {
  for (unsigned k = 2; k <= n; k <<= 1) {
    for (unsigned j = k >> 1; j > 0; j >>= 1) {
      for (unsigned idx = tid; idx < n; idx += nth) {
        unsigned l = idx ^ j;
        if (l > idx) {
          bool dirDesc = ((idx & k) == 0);
          unsigned long long a = keys[idx], c = keys[l];
          if (dirDesc ? (a < c) : (a > c)) { keys[idx] = c; keys[l] = a; }
        }
      }
      __syncthreads();
    }
  }
}

// =====================================================================
// Kernel 3: per-(batch,class) exact top-500 via 4-pass radix select on
// the positive-float bit pattern, then bitonic sort with jax tie-break.
// =====================================================================
__global__ __launch_bounds__(256)
void topk_class_kernel(const float* __restrict__ scores,
                       float* __restrict__ candS, unsigned* __restrict__ candI) {
  const int b   = blockIdx.x / NCLASS;
  const int cls = blockIdx.x % NCLASS;
  const float* sp = scores + (size_t)b * NVOX * NCLASS + cls;

  __shared__ unsigned hist[256];
  __shared__ unsigned s_prefix, s_need, s_cnt;
  __shared__ unsigned long long keys[1024];
  const unsigned tid = threadIdx.x;

  if (tid == 0) { s_prefix = 0u; s_need = KTOP; }
  __syncthreads();

  unsigned maskSoFar = 0u;
  for (int p = 3; p >= 0; --p) {
    for (int j = tid; j < 256; j += 256) hist[j] = 0u;
    __syncthreads();
    const unsigned pref = s_prefix, msk = maskSoFar;
    for (int idx = tid; idx < NVOX; idx += 256) {
      unsigned u = __float_as_uint(sp[(size_t)idx * NCLASS]);
      if ((u & msk) == pref) atomicAdd(&hist[(u >> (8 * p)) & 255u], 1u);
    }
    __syncthreads();
    if (tid == 0) {
      unsigned need = s_need, cum = 0u;
      int bkt = 255;
      for (; bkt > 0; --bkt) {
        if (cum + hist[bkt] >= need) break;
        cum += hist[bkt];
      }
      s_prefix = pref | ((unsigned)bkt << (8 * p));
      s_need = need - cum;
    }
    __syncthreads();
    maskSoFar |= 0xFFu << (8 * p);
  }
  const unsigned T = s_prefix;   // exact bits of the K-th largest value

  if (tid == 0) s_cnt = 0u;
  for (int j = tid; j < 1024; j += 256) keys[j] = 0ull;
  __syncthreads();
  for (int idx = tid; idx < NVOX; idx += 256) {
    unsigned u = __float_as_uint(sp[(size_t)idx * NCLASS]);
    if (u >= T) {
      unsigned pos = atomicAdd(&s_cnt, 1u);
      if (pos < 1024u)
        keys[pos] = ((unsigned long long)u << 32) | (unsigned)(~(unsigned)idx);
    }
  }
  __syncthreads();
  bitonic_desc(keys, 1024, tid, 256);

  for (int r = tid; r < KTOP; r += 256) {
    unsigned long long kv = keys[r];
    candS[(size_t)(b * NCLASS + cls) * KTOP + r] = __uint_as_float((unsigned)(kv >> 32));
    candI[(size_t)(b * NCLASS + cls) * KTOP + r] = ~(unsigned)kv;
  }
}

// =====================================================================
// Kernel 4: per-batch global top-500 of 5000 candidates + box decode.
// key = scorebits<<32 | (8191-flatIdx)<<15 | voxIdx  (jax tie order)
// =====================================================================
__global__ __launch_bounds__(512)
void topk_global_decode_kernel(const float* __restrict__ candS,
                               const unsigned* __restrict__ candI,
                               const float* __restrict__ reg,
                               const int* __restrict__ voxxy,
                               float* __restrict__ out) {
  const int b = blockIdx.x;
  __shared__ unsigned long long keys[8192];
  const unsigned tid = threadIdx.x;

  for (int t = tid; t < 8192; t += 512) {
    unsigned long long kv = 0ull;
    if (t < NCAND) {
      unsigned sb  = __float_as_uint(candS[(size_t)b * NCAND + t]);
      unsigned vox = candI[(size_t)b * NCAND + t] & 0x7FFFu;
      kv = ((unsigned long long)sb << 32)
         | ((unsigned long long)(8191u - (unsigned)t) << 15) | vox;
    }
    keys[t] = kv;
  }
  __syncthreads();
  bitonic_desc(keys, 8192, tid, 512);

  if (tid < KTOP) {
    unsigned long long kv = keys[tid];
    float score   = __uint_as_float((unsigned)(kv >> 32));
    unsigned low  = (unsigned)kv;
    unsigned vox  = low & 0x7FFFu;
    unsigned flat = 8191u - ((low >> 15) & 0x1FFFu);
    int cls = (int)(flat / KTOP);

    const size_t g  = (size_t)b * NVOX + vox;
    const float* rg = reg + g * 10;
    float ctx = rg[0], cty = rg[1], cz = rg[2];
    float d0 = rg[3], d1 = rg[4], d2 = rg[5];
    float rc = rg[6], rs = rg[7], v0 = rg[8], v1 = rg[9];
    float x = (float)voxxy[g * 2 + 0], y = (float)voxxy[g * 2 + 1];

    float xs  = (x + ctx) * 0.6f - 54.0f;   // STRIDE*VOX + PCR
    float ys  = (y + cty) * 0.6f - 54.0f;
    float ang = atan2f(rs, rc);
    bool m = (xs >= -61.2f) && (xs <= 61.2f) && (ys >= -61.2f) && (ys <= 61.2f)
          && (cz >= -10.0f) && (cz <= 10.0f) && (score > 0.1f);
    float mf = m ? 1.0f : 0.0f;

    float box[10] = {xs, ys, cz, d0, d1, d2, ang, v0, v1, score};
    float* ob = out + ((size_t)b * KTOP + tid) * 10;
#pragma unroll
    for (int j = 0; j < 10; ++j) ob[j] = box[j] * mf;
    out[20000 + b * KTOP + tid] = m ? (float)(cls + 1) : 0.0f;  // labels
    out[22000 + b * KTOP + tid] = (float)(vox + b * NVOX);      // voxel_ids
    out[24000 + b * KTOP + tid] = mf;                           // mask
  }
}

// =====================================================================
// Launcher
// =====================================================================
extern "C" void kernel_launch(void* const* d_in, const int* in_sizes, int n_in,
                              void* d_out, int out_size, void* d_ws, size_t ws_size,
                              hipStream_t stream) {
  (void)in_sizes; (void)n_in; (void)out_size; (void)ws_size;
  const float* feats = (const float*)d_in[0];
  const int*   voxxy = (const int*)d_in[1];
  const float* W1  = (const float*)d_in[2];
  const float* b1  = (const float*)d_in[3];
  const float* gma = (const float*)d_in[4];
  const float* bta = (const float*)d_in[5];
  const float* mu  = (const float*)d_in[6];
  const float* var = (const float*)d_in[7];
  const float* W2hm = (const float*)d_in[8];  const float* b2hm = (const float*)d_in[9];
  const float* W2ct = (const float*)d_in[10]; const float* b2ct = (const float*)d_in[11];
  const float* W2cz = (const float*)d_in[12]; const float* b2cz = (const float*)d_in[13];
  const float* W2dm = (const float*)d_in[14]; const float* b2dm = (const float*)d_in[15];
  const float* W2rt = (const float*)d_in[16]; const float* b2rt = (const float*)d_in[17];
  const float* W2vl = (const float*)d_in[18]; const float* b2vl = (const float*)d_in[19];

  char* ws = (char*)d_ws;
  bf16_t* BpHi = (bf16_t*)(ws + OFF_BPHI);
  bf16_t* BpLo = (bf16_t*)(ws + OFF_BPLO);
  float* bnA = (float*)(ws + OFF_BNA);
  float* bnB = (float*)(ws + OFF_BNB);
  float* sc  = (float*)(ws + OFF_SCORES);
  float* rg  = (float*)(ws + OFF_REG);
  float* csc = (float*)(ws + OFF_CSC);
  unsigned* cix = (unsigned*)(ws + OFF_CIX);

  prep_kernel<<<384, 256, 0, stream>>>(W1, b1, gma, bta, mu, var, BpHi, BpLo, bnA, bnB);
  head_gemm_kernel<<<NTOT / 128, 256, 0, stream>>>(
      feats, BpHi, BpLo, bnA, bnB,
      W2hm, b2hm, W2ct, b2ct, W2cz, b2cz, W2dm, b2dm, W2rt, b2rt, W2vl, b2vl,
      sc, rg);
  topk_class_kernel<<<BATCH * NCLASS, 256, 0, stream>>>(sc, csc, cix);
  topk_global_decode_kernel<<<BATCH, 512, 0, stream>>>(csc, cix, rg, voxxy, (float*)d_out);
}
